// ConditionalTimeGANGenerator_32753420599792
// MI455X (gfx1250) — matrix operand order repaired and checked by reading, not compiled, not run on hardware
//
#include <hip/hip_runtime.h>

// ---------------------------------------------------------------------------
// ConditionalTimeGAN generator for MI455X (gfx1250), wave32 + WMMA.
// B=512, D_IN=D_OUT=128, H=512, L=3, E=128, T=256.
// 32 persistent workgroups (one 16-row batch tile each); all T steps run
// in-kernel with hidden state in LDS. GEMMs use v_wmma_f32_16x16x32_f16 with
// f16 weights pre-packed into d_ws (L2-resident, ~9.3MB << 192MB L2).
// 512 threads/block = 16 wave32s = 4 waves per SIMD32, so the L2 latency of
// the B-operand streams is hidden by hardware multithreading; K-loops stay
// rolled to avoid the VGPR-spill storm seen in round 1.
// ---------------------------------------------------------------------------

typedef _Float16 v16h __attribute__((ext_vector_type(16)));
typedef _Float16 h8   __attribute__((ext_vector_type(8)));
typedef float    v8f  __attribute__((ext_vector_type(8)));

#define HD 512      // hidden dim
#define TSTEPS 256
#define BTILE 16    // batch rows per workgroup
#define NTHREADS 512
#define NWAVES 16

// f16 workspace layout (element offsets)
#define OFF_WIH0 0u                 // [1536][256]
#define OFF_WHH0 393216u            // [1536][512]
#define OFF_WIHR 1179648u           // [2][1536][512]
#define OFF_WHHR 2752512u           // [2][1536][512]
#define OFF_W1T  4325376u           // [512][512]  (w1 transposed: [n][k])
#define OFF_W2T  4587520u           // [128][512]  (w2 transposed: [n][k])
#define WS_HALF_TOTAL 4653056u

__device__ __forceinline__ v8f wmma16(v16h a, v16h b, v8f c) {
  return __builtin_amdgcn_wmma_f32_16x16x32_f16(
      /*neg_a=*/false, a, /*neg_b=*/false, b,
      /*c_mod=*/(short)0, c, /*reuse_a=*/false, /*reuse_b=*/false);
}

// A-matrix fragment (16x32 f16, lane m = lane&15, half = lane>>4):
// K chunks [half*8, half*8+8) and [16+half*8, 16+half*8+8)  (ISA 7.12.2)
__device__ __forceinline__ v16h load_fragA(const _Float16* row, int kk, int half) {
  const _Float16* p = row + kk + half * 8;
  h8 c0 = *(const h8*)(p);
  h8 c1 = *(const h8*)(p + 16);
  return __builtin_shufflevector(c0, c1, 0, 1, 2, 3, 4, 5, 6, 7,
                                 8, 9, 10, 11, 12, 13, 14, 15);
}

// B-matrix fragment (32x16 f16, lane n = lane&15): same K order as A,
// K chunks [half*8, half*8+8) and [16+half*8, 16+half*8+8) of the window.
__device__ __forceinline__ v16h load_fragB(const _Float16* row, int kk, int half) {
  const _Float16* p = row + kk + half * 8;
  h8 c0 = *(const h8*)(p);
  h8 c1 = *(const h8*)(p + 16);
  return __builtin_shufflevector(c0, c1, 0, 1, 2, 3, 4, 5, 6, 7,
                                 8, 9, 10, 11, 12, 13, 14, 15);
}

__device__ __forceinline__ float sigm(float x) {
  return 1.0f / (1.0f + __expf(-x));
}
__device__ __forceinline__ float tanh_fast(float x) {
  x = fminf(fmaxf(x, -15.0f), 15.0f);
  float e = __expf(2.0f * x);
  return (e - 1.0f) / (e + 1.0f);
}

// One GRU layer for this block's 16-row tile. Reads old h from h_l (LDS),
// writes new h in place (staged through registers; barrier-protected so the
// full-row gh GEMM reads finish before any wave overwrites h).
__device__ __forceinline__ void gru_layer(
    const _Float16* x_in, int xw,          // LDS input activations [16][xw]
    _Float16* h_l,                         // LDS hidden state [16][512]
    const _Float16* __restrict__ w_ih,     // f16 [1536][xw]
    const _Float16* __restrict__ w_hh,     // f16 [1536][512]
    const float* __restrict__ b_ih, const float* __restrict__ b_hh,
    int lane) {
  const int wave = (int)(threadIdx.x >> 5);
  const int nIdx = lane & 15;
  const int half = lane >> 4;
  const _Float16* arow = x_in + nIdx * xw;   // A row: batch row m = lane&15
  const _Float16* hrow = h_l + nIdx * HD;
  float hnew[2][8];

#pragma unroll
  for (int ci = 0; ci < 2; ++ci) {
    const int cg = (wave + ci * NWAVES) * 16;  // gate-column tile base (0..511)
    v8f accr = {}, accz = {}, accin = {}, acchn = {};

    // gi contribution: x @ W_ih^T  (rows of W_ih are contiguous along K)
    const _Float16* wr = w_ih + (size_t)(cg + nIdx) * xw;
    const _Float16* wz = w_ih + (size_t)(512 + cg + nIdx) * xw;
    const _Float16* wn = w_ih + (size_t)(1024 + cg + nIdx) * xw;
#pragma clang loop unroll(disable)
    for (int kk = 0; kk < xw; kk += 32) {
      v16h a = load_fragA(arow, kk, half);
      accr  = wmma16(a, load_fragB(wr, kk, half), accr);
      accz  = wmma16(a, load_fragB(wz, kk, half), accz);
      accin = wmma16(a, load_fragB(wn, kk, half), accin);
    }
    // gh contribution: h @ W_hh^T
    const _Float16* ur = w_hh + (size_t)(cg + nIdx) * HD;
    const _Float16* uz = w_hh + (size_t)(512 + cg + nIdx) * HD;
    const _Float16* un = w_hh + (size_t)(1024 + cg + nIdx) * HD;
#pragma clang loop unroll(disable)
    for (int kk = 0; kk < HD; kk += 32) {
      v16h a = load_fragA(hrow, kk, half);
      accr  = wmma16(a, load_fragB(ur, kk, half), accr);
      accz  = wmma16(a, load_fragB(uz, kk, half), accz);
      acchn = wmma16(a, load_fragB(un, kk, half), acchn);
    }

    const float bir = b_ih[cg + nIdx], biz = b_ih[512 + cg + nIdx],
                bin = b_ih[1024 + cg + nIdx];
    const float bhr = b_hh[cg + nIdx], bhz = b_hh[512 + cg + nIdx],
                bhn = b_hh[1024 + cg + nIdx];
#pragma unroll
    for (int j = 0; j < 8; ++j) {
      const int m = j + half * 8;            // C/D layout: VGPR j, lane half
      float r  = sigm(accr[j] + bir + bhr);
      float z  = sigm(accz[j] + biz + bhz);
      float nv = tanh_fast(accin[j] + bin + r * (acchn[j] + bhn));
      float ho = (float)h_l[m * HD + cg + nIdx];
      hnew[ci][j] = (1.0f - z) * nv + z * ho;
    }
  }

  __syncthreads();   // all reads of old h complete across all waves
#pragma unroll
  for (int ci = 0; ci < 2; ++ci) {
    const int cg = (wave + ci * NWAVES) * 16;
#pragma unroll
    for (int j = 0; j < 8; ++j)
      h_l[(j + half * 8) * HD + cg + nIdx] = (_Float16)hnew[ci][j];
  }
  __syncthreads();   // new h visible before next consumer
}

__global__ __launch_bounds__(NTHREADS, 1) void tgan_main(
    const float* __restrict__ noise, const int* __restrict__ cond,
    const float* __restrict__ emb, const _Float16* __restrict__ ws,
    const float* __restrict__ b_ih0, const float* __restrict__ b_hh0,
    const float* __restrict__ b_ih_r, const float* __restrict__ b_hh_r,
    const float* __restrict__ b1, const float* __restrict__ b2,
    float* __restrict__ out) {
  __shared__ alignas(16) _Float16 x_lds[BTILE * 256];      // [y | cond_emb]
  __shared__ alignas(16) _Float16 h_lds[3][BTILE * HD];    // hidden states
  __shared__ alignas(16) _Float16 a_lds[BTILE * HD];       // MLP hidden

  const int tid  = (int)threadIdx.x;
  const int lane = tid & 31;
  const int wave = tid >> 5;
  const int nIdx = lane & 15;
  const int half = lane >> 4;
  const int b0   = (int)blockIdx.x * BTILE;

  // init x = [noise | emb[cond]] (f16), h = 0
  for (int i = tid; i < BTILE * 256; i += NTHREADS) {
    int m = i >> 8, c = i & 255;
    float v = (c < 128) ? noise[(size_t)(b0 + m) * 128 + c]
                        : emb[(size_t)cond[b0 + m] * 128 + (c - 128)];
    x_lds[i] = (_Float16)v;
  }
  for (int i = tid; i < 3 * BTILE * HD; i += NTHREADS)
    (&h_lds[0][0])[i] = (_Float16)0.0f;
  __syncthreads();

#pragma clang loop unroll(disable)
  for (int t = 0; t < TSTEPS; ++t) {
    gru_layer(x_lds, 256, h_lds[0], ws + OFF_WIH0, ws + OFF_WHH0,
              b_ih0, b_hh0, lane);
    gru_layer(h_lds[0], HD, h_lds[1], ws + OFF_WIHR, ws + OFF_WHHR,
              b_ih_r, b_hh_r, lane);
    gru_layer(h_lds[1], HD, h_lds[2], ws + OFF_WIHR + 786432u,
              ws + OFF_WHHR + 786432u, b_ih_r + 1536, b_hh_r + 1536, lane);

    // MLP layer 1: a = relu(h2 @ w1 + b1)
    {
      const _Float16* arow = h_lds[2] + nIdx * HD;
#pragma unroll
      for (int ci = 0; ci < 2; ++ci) {
        const int cg = (wave + ci * NWAVES) * 16;
        v8f acc = {};
        const _Float16* wrow = ws + OFF_W1T + (size_t)(cg + nIdx) * HD;
#pragma clang loop unroll(disable)
        for (int kk = 0; kk < HD; kk += 32)
          acc = wmma16(load_fragA(arow, kk, half), load_fragB(wrow, kk, half), acc);
        const float bb = b1[cg + nIdx];
#pragma unroll
        for (int j = 0; j < 8; ++j)
          a_lds[(j + half * 8) * HD + cg + nIdx] =
              (_Float16)fmaxf(acc[j] + bb, 0.0f);
      }
    }
    __syncthreads();

    // MLP layer 2: y = tanh(a @ w2 + b2); emit output + feed back into x.
    // Only 8 column tiles (128 cols) -> waves 0..7 active.
    if (wave < 8) {
      const _Float16* arow = a_lds + nIdx * HD;
      const int cg = wave * 16;
      v8f acc = {};
      const _Float16* wrow = ws + OFF_W2T + (size_t)(cg + nIdx) * HD;
#pragma clang loop unroll(disable)
      for (int kk = 0; kk < HD; kk += 32)
        acc = wmma16(load_fragA(arow, kk, half), load_fragB(wrow, kk, half), acc);
      const float bb = b2[cg + nIdx];
#pragma unroll
      for (int j = 0; j < 8; ++j) {
        const int m = j + half * 8;
        float y = tanh_fast(acc[j] + bb);
        out[((size_t)(b0 + m) * TSTEPS + t) * 128 + cg + nIdx] = y;
        x_lds[m * 256 + cg + nIdx] = (_Float16)y;
      }
    }
    __syncthreads();
  }
}

// --- weight pre-pack kernels: fp32 -> f16 (w1/w2 also transposed to [n][k]) --
__global__ void cvt_copy(const float* __restrict__ s, _Float16* __restrict__ d,
                         unsigned n) {
  unsigned stride = gridDim.x * blockDim.x;
  for (unsigned i = blockIdx.x * blockDim.x + threadIdx.x; i < n; i += stride)
    d[i] = (_Float16)s[i];
}
__global__ void cvt_transpose(const float* __restrict__ s,
                              _Float16* __restrict__ d, int R, int C) {
  // src [R][C] row-major -> dst [C][R]
  unsigned n = (unsigned)R * (unsigned)C;
  unsigned stride = gridDim.x * blockDim.x;
  for (unsigned i = blockIdx.x * blockDim.x + threadIdx.x; i < n; i += stride) {
    unsigned nn = i / (unsigned)R, k = i % (unsigned)R;
    d[i] = (_Float16)s[k * (unsigned)C + nn];
  }
}

extern "C" void kernel_launch(void* const* d_in, const int* in_sizes, int n_in,
                              void* d_out, int out_size, void* d_ws,
                              size_t ws_size, hipStream_t stream) {
  (void)in_sizes; (void)n_in; (void)out_size;
  if (ws_size < (size_t)WS_HALF_TOTAL * sizeof(_Float16)) return;

  const float* noise  = (const float*)d_in[0];
  const int*   cond   = (const int*)d_in[1];
  // d_in[2] = seq_len (T=256, fixed by the problem instance)
  const float* emb    = (const float*)d_in[3];
  const float* w_ih0  = (const float*)d_in[4];
  const float* w_hh0  = (const float*)d_in[5];
  const float* b_ih0  = (const float*)d_in[6];
  const float* b_hh0  = (const float*)d_in[7];
  const float* w_ih_r = (const float*)d_in[8];
  const float* w_hh_r = (const float*)d_in[9];
  const float* b_ih_r = (const float*)d_in[10];
  const float* b_hh_r = (const float*)d_in[11];
  const float* w1     = (const float*)d_in[12];
  const float* b1     = (const float*)d_in[13];
  const float* w2     = (const float*)d_in[14];
  const float* b2     = (const float*)d_in[15];
  _Float16* ws = (_Float16*)d_ws;

  cvt_copy<<<512, 256, 0, stream>>>(w_ih0, ws + OFF_WIH0, 1536u * 256u);
  cvt_copy<<<512, 256, 0, stream>>>(w_hh0, ws + OFF_WHH0, 1536u * 512u);
  cvt_copy<<<1024, 256, 0, stream>>>(w_ih_r, ws + OFF_WIHR, 2u * 1536u * 512u);
  cvt_copy<<<1024, 256, 0, stream>>>(w_hh_r, ws + OFF_WHHR, 2u * 1536u * 512u);
  cvt_transpose<<<512, 256, 0, stream>>>(w1, ws + OFF_W1T, 512, 512);
  cvt_transpose<<<128, 256, 0, stream>>>(w2, ws + OFF_W2T, 512, 128);

  tgan_main<<<512 / BTILE, NTHREADS, 0, stream>>>(
      noise, cond, emb, ws, b_ih0, b_hh0, b_ih_r, b_hh_r, b1, b2,
      (float*)d_out);
}
